// _Rank2DecompositionEdgeBlock_7808250544508
// MI455X (gfx1250) — matrix operand order repaired and checked by reading, not compiled, not run on hardware
//
#include <hip/hip_runtime.h>
#include <hip/hip_bf16.h>
#include <math.h>

// ---------------------------------------------------------------------------
// Types for CDNA5 WMMA (wave32): D(16x16 f32) = A(16x32 bf16) x B(32x16 bf16) + C
// ---------------------------------------------------------------------------
typedef __attribute__((ext_vector_type(16))) __bf16 v16bf;
typedef __attribute__((ext_vector_type(8)))  __bf16 v8bf;
typedef __attribute__((ext_vector_type(8)))  float  v8f;

union BFrag {            // B fragment: two 16-byte LDS loads
    v16bf v;
    v8bf  h[2];
};
union AFrag {            // A fragment: element-wise fill from converted f32
    v16bf  v;
    __bf16 e[16];
};

#define EMB        256
#define WPB        8            // waves per block
#define MT         2            // 16-edge M-tiles per wave (32 edges/wave)
#define ROWSTRIDE  264          // 256 + 8 bf16 pad -> conflict-free ds_load_b128
#define S3F        1.7320508075688772f
#define SH2SCALE   0.6307831305050401f   // sqrt(5/(4*pi))

// ---------------------------------------------------------------------------
// Zero workspace accumulators
// ---------------------------------------------------------------------------
__global__ void zero_ws_kernel(float* __restrict__ p, int n) {
    int i = blockIdx.x * blockDim.x + threadIdx.x;
    if (i < n) p[i] = 0.0f;
}

// ---------------------------------------------------------------------------
// Fused: dual MLP (shared input) via bf16 WMMA + SiLU + second layer dot +
//        sph-harmonics weighting + atomic scatter to node accumulators.
// nodeAcc layout per node: [scalar, irr0..irr4, count]  (7 floats)
// ---------------------------------------------------------------------------
__global__ __launch_bounds__(256, 1)
void edge_mlp_wmma_kernel(const float* __restrict__ x_edge,
                          const float* __restrict__ edge_vec,
                          const float* __restrict__ s_w1,
                          const float* __restrict__ s_b1,
                          const float* __restrict__ s_w2,
                          const float* __restrict__ s_b2,
                          const float* __restrict__ i_w1,
                          const float* __restrict__ i_b1,
                          const float* __restrict__ i_w2,
                          const float* __restrict__ i_b2,
                          const int*   __restrict__ idx_t,
                          float*       __restrict__ nodeAcc,
                          int numTiles)
{
    extern __shared__ char smem[];
    __bf16* wS  = (__bf16*)smem;                                   // 256 x 264 bf16
    __bf16* wI  = wS + EMB * ROWSTRIDE;                            // 256 x 264 bf16
    float*  red = (float*)(smem + (size_t)2 * EMB * ROWSTRIDE * 2);// 8 waves x 64 f32

    const int tid  = threadIdx.x;
    const int lane = tid & 31;
    const int wv   = tid >> 5;

    // ---- Stage both weight matrices into LDS as bf16 (row-padded) ----
    for (int i = tid * 4; i < EMB * EMB; i += 256 * 4) {
        int j = i >> 8;          // row (output neuron)
        int k = i & 255;         // col (input feature)
        float4 a = *(const float4*)(s_w1 + i);
        __bf16* ds = wS + (size_t)j * ROWSTRIDE + k;
        ds[0] = (__bf16)a.x; ds[1] = (__bf16)a.y; ds[2] = (__bf16)a.z; ds[3] = (__bf16)a.w;
        float4 b = *(const float4*)(i_w1 + i);
        __bf16* di = wI + (size_t)j * ROWSTRIDE + k;
        di[0] = (__bf16)b.x; di[1] = (__bf16)b.y; di[2] = (__bf16)b.z; di[3] = (__bf16)b.w;
    }
    __syncthreads();

    const int  tile   = blockIdx.x * WPB + wv;   // 32-edge tile
    const bool active = tile < numTiles;
    const int  edge0  = tile * (16 * MT);
    const int  khalf  = lane >> 4;        // 0: lanes 0-15, 1: lanes 16-31
    const int  jlane  = lane & 15;

    float paccS[MT][8], paccI[MT][8];
    #pragma unroll
    for (int t = 0; t < MT; ++t)
        #pragma unroll
        for (int r = 0; r < 8; ++r) { paccS[t][r] = 0.0f; paccI[t][r] = 0.0f; }

    if (active) {
        // ---- Load A fragments (16x32 bf16 per k-step) once, keep in regs ----
        // Lane l holds M = l%16; K runs {base..base+7} and {base+16..base+23},
        // base = (l/16)*8  (ISA 7.12.2 16-bit A layout).
        AFrag afrag[MT][8];
        const int abase = khalf * 8;
        #pragma unroll
        for (int t = 0; t < MT; ++t) {
            const float* xrow = x_edge + (size_t)(edge0 + t * 16 + jlane) * EMB;
            #pragma unroll
            for (int kk = 0; kk < 8; ++kk) {
                const float4* p0 = (const float4*)(xrow + kk * 32 + abase);
                float4 a0 = p0[0], a1 = p0[1];
                const float4* p1 = (const float4*)(xrow + kk * 32 + abase + 16);
                float4 a2 = p1[0], a3 = p1[1];
                AFrag f;
                f.e[0]  = (__bf16)a0.x; f.e[1]  = (__bf16)a0.y;
                f.e[2]  = (__bf16)a0.z; f.e[3]  = (__bf16)a0.w;
                f.e[4]  = (__bf16)a1.x; f.e[5]  = (__bf16)a1.y;
                f.e[6]  = (__bf16)a1.z; f.e[7]  = (__bf16)a1.w;
                f.e[8]  = (__bf16)a2.x; f.e[9]  = (__bf16)a2.y;
                f.e[10] = (__bf16)a2.z; f.e[11] = (__bf16)a2.w;
                f.e[12] = (__bf16)a3.x; f.e[13] = (__bf16)a3.y;
                f.e[14] = (__bf16)a3.z; f.e[15] = (__bf16)a3.w;
                afrag[t][kk] = f;
            }
        }

        // ---- 16 hidden tiles of 16; 8 WMMA k-steps each; both MLPs;
        //      B fragments double-buffered so loads overlap WMMA ----
        for (int jt = 0; jt < 16; ++jt) {
            const int j = jt * 16 + jlane;     // hidden unit for this lane's N
            const size_t rowoff = (size_t)j * ROWSTRIDE;
            v8f accS0 = {0.f,0.f,0.f,0.f,0.f,0.f,0.f,0.f};
            v8f accS1 = {0.f,0.f,0.f,0.f,0.f,0.f,0.f,0.f};
            v8f accI0 = {0.f,0.f,0.f,0.f,0.f,0.f,0.f,0.f};
            v8f accI1 = {0.f,0.f,0.f,0.f,0.f,0.f,0.f,0.f};

            BFrag bs, bi;
            {   // prologue: k-step 0
                const int koff = khalf * 8;
                const v8bf* ps = (const v8bf*)(wS + rowoff + koff);
                bs.h[0] = ps[0]; bs.h[1] = ps[2];
                const v8bf* pi = (const v8bf*)(wI + rowoff + koff);
                bi.h[0] = pi[0]; bi.h[1] = pi[2];
            }
            #pragma unroll
            for (int kk = 0; kk < 8; ++kk) {
                BFrag nbs, nbi;
                if (kk < 7) {   // prefetch next k-step while WMMAs run
                    const int koff = (kk + 1) * 32 + khalf * 8;
                    const v8bf* ps = (const v8bf*)(wS + rowoff + koff);
                    nbs.h[0] = ps[0]; nbs.h[1] = ps[2];
                    const v8bf* pi = (const v8bf*)(wI + rowoff + koff);
                    nbi.h[0] = pi[0]; nbi.h[1] = pi[2];
                }
                accS0 = __builtin_amdgcn_wmma_f32_16x16x32_bf16(
                        false, afrag[0][kk].v, false, bs.v, (short)0, accS0, false, false);
                accS1 = __builtin_amdgcn_wmma_f32_16x16x32_bf16(
                        false, afrag[1][kk].v, false, bs.v, (short)0, accS1, false, false);
                accI0 = __builtin_amdgcn_wmma_f32_16x16x32_bf16(
                        false, afrag[0][kk].v, false, bi.v, (short)0, accI0, false, false);
                accI1 = __builtin_amdgcn_wmma_f32_16x16x32_bf16(
                        false, afrag[1][kk].v, false, bi.v, (short)0, accI1, false, false);
                if (kk < 7) { bs = nbs; bi = nbi; }
            }

            // bias + SiLU + dot with second-layer weights (f32 VALU)
            const float b1s = s_b1[j], b1i = i_b1[j];
            const float w2s = s_w2[j], w2i = i_w2[j];
            #pragma unroll
            for (int r = 0; r < 8; ++r) {
                float hs0 = accS0[r] + b1s;
                paccS[0][r] += (hs0 / (1.0f + __expf(-hs0))) * w2s;
                float hs1 = accS1[r] + b1s;
                paccS[1][r] += (hs1 / (1.0f + __expf(-hs1))) * w2s;
                float hi0 = accI0[r] + b1i;
                paccI[0][r] += (hi0 / (1.0f + __expf(-hi0))) * w2i;
                float hi1 = accI1[r] + b1i;
                paccI[1][r] += (hi1 / (1.0f + __expf(-hi1))) * w2i;
            }
        }

        // ---- reduce across the 16 lanes holding the same edge (M) ----
        #pragma unroll
        for (int t = 0; t < MT; ++t) {
            #pragma unroll
            for (int r = 0; r < 8; ++r) {
                float s = paccS[t][r], v = paccI[t][r];
                #pragma unroll
                for (int m = 1; m < 16; m <<= 1) {
                    s += __shfl_xor(s, m, 32);
                    v += __shfl_xor(v, m, 32);
                }
                if (jlane == r) {
                    int idx = t * 16 + r + (khalf << 3);
                    red[wv * 64 + idx]      = s;
                    red[wv * 64 + 32 + idx] = v;
                }
            }
        }
    }
    __syncthreads();

    // ---- tail: sph harmonics + atomic scatter, one edge per lane ----
    if (active) {
        const int   e  = edge0 + lane;
        const float sv = red[wv * 64 + lane]      + s_b2[0];
        const float iv = red[wv * 64 + 32 + lane] + i_b2[0];

        const float vx = edge_vec[(size_t)e * 3 + 0];
        const float vy = edge_vec[(size_t)e * 3 + 1];
        const float vz = edge_vec[(size_t)e * 3 + 2];
        const float inv = rsqrtf(vx * vx + vy * vy + vz * vz);
        const float x = vx * inv, y = vy * inv, z = vz * inv;

        const float sh0 = (S3F * x * z) * SH2SCALE;
        const float sh1 = (S3F * x * y) * SH2SCALE;
        const float sh2 = (y * y - 0.5f * (x * x + z * z)) * SH2SCALE;
        const float sh3 = (S3F * y * z) * SH2SCALE;
        const float sh4 = (0.5f * S3F * (z * z - x * x)) * SH2SCALE;

        const int node = idx_t[e];
        float* p = nodeAcc + (size_t)node * 7;
        atomicAdd(p + 0, sv);
        atomicAdd(p + 1, sh0 * iv);
        atomicAdd(p + 2, sh1 * iv);
        atomicAdd(p + 3, sh2 * iv);
        atomicAdd(p + 4, sh3 * iv);
        atomicAdd(p + 5, sh4 * iv);
        atomicAdd(p + 6, 1.0f);
    }
}

// ---------------------------------------------------------------------------
// nodes -> batch segment mean accumulation
// batchAcc layout per batch: [scalar, irr0..irr4, count, pad]  (8 floats)
// ---------------------------------------------------------------------------
__global__ void node_to_batch_kernel(const float* __restrict__ nodeAcc,
                                     const int*   __restrict__ batch_idx,
                                     float*       __restrict__ batchAcc,
                                     int N)
{
    int n = blockIdx.x * blockDim.x + threadIdx.x;
    if (n >= N) return;
    const float* p = nodeAcc + (size_t)n * 7;
    float cnt = p[6];
    cnt = cnt < 1.0f ? 1.0f : cnt;
    const int b = batch_idx[n];
    float* q = batchAcc + (size_t)b * 8;
    atomicAdd(q + 0, p[0] / cnt);
    #pragma unroll
    for (int c = 0; c < 5; ++c) atomicAdd(q + 1 + c, p[1 + c] / cnt);
    atomicAdd(q + 6, 1.0f);
}

// ---------------------------------------------------------------------------
// finalize: batch means + 9x9 change-of-basis -> (B,3,3)
// out[c*9+a] = sum_b M[b][a] * flat[c][b], flat=[scalar,0,0,0,irr0..irr4]
// ---------------------------------------------------------------------------
__global__ void finalize_kernel(const float* __restrict__ batchAcc,
                                float*       __restrict__ out,
                                int B)
{
    const float R2 = 0.7071067811865476f;
    const float R3 = 0.5773502691896258f;   // 3^-1/2
    const float R6 = 0.4082482904638630f;   // 6^-1/2
    // rows 0,4,5,6,7,8 of _M (rows 1..3 multiply the zero vector part)
    const float M0[9] = { R3, 0.f, 0.f, 0.f,  R3, 0.f, 0.f, 0.f,  R3};
    const float M4[9] = {0.f, 0.f,  R2, 0.f, 0.f, 0.f,  R2, 0.f, 0.f};
    const float M5[9] = {0.f,  R2, 0.f,  R2, 0.f, 0.f, 0.f, 0.f, 0.f};
    const float M6[9] = {-R6, 0.f, 0.f, 0.f, 2.f*R6, 0.f, 0.f, 0.f, -R6};
    const float M7[9] = {0.f, 0.f, 0.f, 0.f, 0.f,  R2, 0.f,  R2, 0.f};
    const float M8[9] = {-R2, 0.f, 0.f, 0.f, 0.f, 0.f, 0.f, 0.f,  R2};

    const int t = threadIdx.x;
    const int c = t / 9, a = t % 9;
    if (c >= B) return;
    const float* q = batchAcc + (size_t)c * 8;
    float cnt = q[6];
    cnt = cnt < 1.0f ? 1.0f : cnt;
    const float s  = q[0] / cnt;
    const float i0 = q[1] / cnt, i1 = q[2] / cnt, i2 = q[3] / cnt;
    const float i3 = q[4] / cnt, i4 = q[5] / cnt;
    out[c * 9 + a] = M0[a] * s + M4[a] * i0 + M5[a] * i1 +
                     M6[a] * i2 + M7[a] * i3 + M8[a] * i4;
}

// ---------------------------------------------------------------------------
extern "C" void kernel_launch(void* const* d_in, const int* in_sizes, int n_in,
                              void* d_out, int out_size, void* d_ws, size_t ws_size,
                              hipStream_t stream) {
    const float* x_edge   = (const float*)d_in[0];
    const float* edge_vec = (const float*)d_in[1];
    const float* s_w1     = (const float*)d_in[2];
    const float* s_b1     = (const float*)d_in[3];
    const float* s_w2     = (const float*)d_in[4];
    const float* s_b2     = (const float*)d_in[5];
    const float* i_w1     = (const float*)d_in[6];
    const float* i_b1     = (const float*)d_in[7];
    const float* i_w2     = (const float*)d_in[8];
    const float* i_b2     = (const float*)d_in[9];
    const int*   idx_t    = (const int*)d_in[10];
    const int*   batch_i  = (const int*)d_in[11];

    const int E = in_sizes[1] / 3;          // 500000
    const int N = in_sizes[11];             // 20000
    const int B = 16;

    float* nodeAcc  = (float*)d_ws;                 // N*7 floats
    float* batchAcc = nodeAcc + (size_t)N * 7;      // B*8 floats

    // zero accumulators
    const int zn = N * 7 + B * 8;
    zero_ws_kernel<<<(zn + 255) / 256, 256, 0, stream>>>(nodeAcc, zn);

    // fused WMMA edge kernel: each wave owns 32 edges (2 M-tiles)
    const int numTiles = E / (16 * MT);
    const int blocks   = (numTiles + WPB - 1) / WPB;
    const size_t smem  = (size_t)2 * EMB * ROWSTRIDE * sizeof(__bf16)
                       + (size_t)WPB * 64 * sizeof(float);
    hipFuncSetAttribute((const void*)edge_mlp_wmma_kernel,
                        hipFuncAttributeMaxDynamicSharedMemorySize, (int)smem);
    edge_mlp_wmma_kernel<<<blocks, 256, smem, stream>>>(
        x_edge, edge_vec, s_w1, s_b1, s_w2, s_b2,
        i_w1, i_b1, i_w2, i_b2, idx_t, nodeAcc, numTiles);

    // node -> batch segment means
    node_to_batch_kernel<<<(N + 255) / 256, 256, 0, stream>>>(
        nodeAcc, batch_i, batchAcc, N);

    // batch means + change of basis
    finalize_kernel<<<1, 160, 0, stream>>>(batchAcc, (float*)d_out, B);
}
